// ChebNet_34299608826051
// MI455X (gfx1250) — compile-verified, matrix operand order checked by reading, not run on hardware
//
#include <hip/hip_runtime.h>
#include <hip/hip_bf16.h>

// ---------------------------------------------------------------------------
// MI455X plan (unchanged from round 1):
//  - Dominant cost: adjacency MLP = GEMM with M=1024*676, K=F(23/55/87),
//    N=128 -> 64 -> 1 (~63 GFLOP), compute-bound -> v_wmma_f32_16x16x32_f16
//    with BN folded into the linears and weights pre-swizzled into
//    B-fragment order (one contiguous 32B load per lane, L2-broadcast).
//  - Round-2 improvements: zero-padded f16 feature staging kills the
//    exec-masked per-element guards; A fragments are built from two
//    ds_load_b128 runs per k-step (the 16-bit A layout is two contiguous
//    8-half runs per lane) with packed-f16 |a-b|; final dot vectorized.
// ---------------------------------------------------------------------------

typedef __attribute__((ext_vector_type(16))) _Float16 v16h;
typedef __attribute__((ext_vector_type(8)))  _Float16 v8h;
typedef __attribute__((ext_vector_type(8)))  float    v8f;
typedef __attribute__((ext_vector_type(4)))  unsigned int v4u;

#define BSZ   1024
#define NNOD  26
#define NPAIR 676       // 26*26
#define MT    43        // ceil(676/16)
#define HPITCH 96       // feature pitch per node in workspace

// |x| on 8 packed halves via bit-AND
static __device__ __forceinline__ v8h habs8(v8h x) {
    v4u u;
    __builtin_memcpy(&u, &x, 16);
    u &= 0x7FFF7FFFu;
    v8h r;
    __builtin_memcpy(&r, &u, 16);
    return r;
}

// ---------------------------------------------------------------------------
// Fold CNN batchnorms + conv biases: out = conv*a + d
// fold layout: a1[32] d1[32] a2 d2 a3 d3 a4 d4  (256 floats)
// ---------------------------------------------------------------------------
__global__ void prep_cnn(const float* c1b, const float* g1, const float* be1, const float* m1, const float* v1,
                         const float* c2b, const float* g2, const float* be2, const float* m2, const float* v2,
                         const float* c3b, const float* g3, const float* be3, const float* m3, const float* v3,
                         const float* f1b, const float* g4, const float* be4, const float* m4, const float* v4,
                         float* fold) {
    int t = threadIdx.x;
    if (t < 32) {
        float s;
        s = g1[t] * rsqrtf(v1[t] + 1e-5f); fold[t]       = s; fold[32+t]  = (c1b[t]-m1[t])*s + be1[t];
        s = g2[t] * rsqrtf(v2[t] + 1e-5f); fold[64+t]    = s; fold[96+t]  = (c2b[t]-m2[t])*s + be2[t];
        s = g3[t] * rsqrtf(v3[t] + 1e-5f); fold[128+t]   = s; fold[160+t] = (c3b[t]-m3[t])*s + be3[t];
        s = g4[t] * rsqrtf(v4[t] + 1e-5f); fold[192+t]   = s; fold[224+t] = (f1b[t]-m4[t])*s + be4[t];
    }
}

// ---------------------------------------------------------------------------
// CNN feature extractor: one block (64 threads) per sample (26624 samples).
// Writes H[b][node][0..17] = features, H[b][node][18..22] = label one-hot/pad.
// ---------------------------------------------------------------------------
__global__ void __launch_bounds__(64) cnn_fwd(const float* __restrict__ x, const float* __restrict__ xi,
                                              const float* __restrict__ oh,
                                              const float* __restrict__ c1w, const float* __restrict__ c2w,
                                              const float* __restrict__ c3w,
                                              const float* __restrict__ f1w, const float* __restrict__ f2w,
                                              const float* __restrict__ fold, const float* __restrict__ f2b,
                                              float* __restrict__ H) {
    int s = blockIdx.x;             // 0..26623
    int b = s / NNOD, r = s % NNOD; // node r (0 = query x, 1..25 = support xi)
    const float* in = (r == 0) ? (x + (size_t)b * 120)
                               : (xi + ((size_t)b * 25 + (r - 1)) * 120);
    __shared__ float in_s[120];
    __shared__ float h1[32][60];
    __shared__ float h2[32][30];
    __shared__ float h3[480];
    __shared__ float h4[32];
    int t = threadIdx.x;
    for (int i = t; i < 120; i += 64) in_s[i] = in[i];
    __syncthreads();
    // conv1 (1->32, k5, s2, p2): out 60
    for (int idx = t; idx < 32 * 60; idx += 64) {
        int c = idx / 60, pp = idx % 60;
        float acc = 0.f;
#pragma unroll
        for (int k = 0; k < 5; ++k) {
            int q = 2 * pp + k - 2;
            if (q >= 0 && q < 120) acc += c1w[c * 5 + k] * in_s[q];
        }
        acc = acc * fold[c] + fold[32 + c];
        h1[c][pp] = fmaxf(acc, 0.f);
    }
    __syncthreads();
    // conv2 (32->32): out 30
    for (int idx = t; idx < 32 * 30; idx += 64) {
        int c = idx / 30, pp = idx % 30;
        float acc = 0.f;
        for (int ci = 0; ci < 32; ++ci) {
            const float* w = c2w + (c * 32 + ci) * 5;
#pragma unroll
            for (int k = 0; k < 5; ++k) {
                int q = 2 * pp + k - 2;
                if (q >= 0 && q < 60) acc += w[k] * h1[ci][q];
            }
        }
        acc = acc * fold[64 + c] + fold[96 + c];
        h2[c][pp] = fmaxf(acc, 0.f);
    }
    __syncthreads();
    // conv3 (32->32): out 15
    for (int idx = t; idx < 32 * 15; idx += 64) {
        int c = idx / 15, pp = idx % 15;
        float acc = 0.f;
        for (int ci = 0; ci < 32; ++ci) {
            const float* w = c3w + (c * 32 + ci) * 5;
#pragma unroll
            for (int k = 0; k < 5; ++k) {
                int q = 2 * pp + k - 2;
                if (q >= 0 && q < 30) acc += w[k] * h2[ci][q];
            }
        }
        acc = acc * fold[128 + c] + fold[160 + c];
        h3[c * 15 + pp] = fmaxf(acc, 0.f);
    }
    __syncthreads();
    // f1 (480x32) + bn4 + relu
    if (t < 32) {
        float acc = 0.f;
        for (int f = 0; f < 480; ++f) acc += h3[f] * f1w[f * 32 + t];
        acc = acc * fold[192 + t] + fold[224 + t];
        h4[t] = fmaxf(acc, 0.f);
    }
    __syncthreads();
    float* Hrow = H + ((size_t)b * NNOD + r) * HPITCH;
    if (t < 18) {
        float acc = f2b[t];
        for (int f = 0; f < 32; ++f) acc += h4[f] * f2w[f * 18 + t];
        Hrow[t] = acc;
    }
    if (t >= 18 && t < 23) {
        int c = t - 18;
        Hrow[t] = (r == 0) ? 0.2f : oh[((size_t)b * 25 + (r - 1)) * 5 + c];
    }
}

// ---------------------------------------------------------------------------
// B-fragment K pattern per the ISA 16-bit matrix layout: for lane L, elem e,
// k = 8*(L/16)+e (e<8) or 16+8*(L/16)+(e-8) (e>=8)  -- two contiguous runs.
// ---------------------------------------------------------------------------
__device__ __forceinline__ int kpat(int lane, int e) {
    int half = lane >> 4;
    return (e < 8) ? (8 * half + e) : (16 + 8 * half + (e - 8));
}

template <int F, int KS0>
__global__ void __launch_bounds__(256) prep_adj(
    const float* w0, const float* b0l, const float* g0, const float* be0, const float* m0, const float* v0,
    const float* w1, const float* b1l, const float* g1, const float* be1, const float* m1, const float* v1,
    const float* w2, const float* b2,
    _Float16* W0sw, float* bias0, _Float16* W1sw, float* bias1, float* w2f) {
    int t = threadIdx.x;
    for (int n = t; n < 128; n += 256) {
        float sc = g0[n] * rsqrtf(v0[n] + 1e-5f);
        bias0[n] = (b0l[n] - m0[n]) * sc + be0[n];
    }
    for (int n = t; n < 64; n += 256) {
        float sc = g1[n] * rsqrtf(v1[n] + 1e-5f);
        bias1[n] = (b1l[n] - m1[n]) * sc + be1[n];
        w2f[n] = w2[n];
    }
    if (t == 0) w2f[64] = b2[0];
    // W0: (F x 128) -> 8 n-tiles x KS0 k-steps
    for (int idx = t; idx < 8 * KS0 * 512; idx += 256) {
        int frag = idx >> 9, within = idx & 511;
        int lane = within >> 4, e = within & 15;
        int nt = frag / KS0, ks = frag % KS0;
        int n = nt * 16 + (lane & 15);
        int k = ks * 32 + kpat(lane, e);
        float sc = g0[n] * rsqrtf(v0[n] + 1e-5f);
        float w = (k < F) ? w0[k * 128 + n] * sc : 0.f;
        W0sw[idx] = (_Float16)w;
    }
    // W1: (128 x 64) -> 4 n-tiles x 4 k-steps
    for (int idx = t; idx < 16 * 512; idx += 256) {
        int frag = idx >> 9, within = idx & 511;
        int lane = within >> 4, e = within & 15;
        int nt = frag >> 2, ks = frag & 3;
        int n = nt * 16 + (lane & 15);
        int k = ks * 32 + kpat(lane, e);
        float sc = g1[n] * rsqrtf(v1[n] + 1e-5f);
        W1sw[idx] = (_Float16)(w1[k * 64 + n] * sc);
    }
}

// ---------------------------------------------------------------------------
// Fused adjacency MLP: per 16-pair tile, phi(16xF) @ W0 -> leaky ->
// (16x128) @ W1 -> leaky -> (16x64) . w2 + b2 -> s scores. WMMA f16/f32.
// One block per episode b (8 waves x 43 tiles). EXEC is full everywhere a
// WMMA issues (wave-uniform control flow, no divergent guards).
// ---------------------------------------------------------------------------
template <int F, int KS0>
__global__ void __launch_bounds__(256) adj_mlp(
    const float* __restrict__ Hbuf,
    const _Float16* __restrict__ W0sw, const float* __restrict__ bias0,
    const _Float16* __restrict__ W1sw, const float* __restrict__ bias1,
    const float* __restrict__ w2f,
    float* __restrict__ sbuf) {
    constexpr int FP = KS0 * 32;       // padded K
    constexpr int HSP = FP + 8;        // Hs pitch (halves, mult of 8 -> 16B)
    const int b = blockIdx.x;
    const int wave = threadIdx.x >> 5;
    const int lane = threadIdx.x & 31;
    const int half = lane >> 4;
    const int ml = lane & 15;

    __shared__ _Float16 Hs[NNOD][HSP];         // zero-padded f16 node features
    __shared__ _Float16 h0s[8][16][136];       // per-wave hidden tile bounce
    __shared__ float b0s[128], b1s[64], w2s[65];

    for (int idx = threadIdx.x; idx < NNOD * FP; idx += 256) {
        int n = idx / FP, f = idx % FP;
        float v = (f < F) ? Hbuf[((size_t)b * NNOD + n) * HPITCH + f] : 0.f;
        Hs[n][f] = (_Float16)v;
    }
    for (int i = threadIdx.x; i < 128; i += 256) b0s[i] = bias0[i];
    for (int i = threadIdx.x; i < 64; i += 256) { b1s[i] = bias1[i]; w2s[i] = w2f[i]; }
    if (threadIdx.x == 0) w2s[64] = w2f[64];
    __syncthreads();

    for (int tile = wave; tile < MT; tile += 8) {
        const int p0 = tile * 16;
        const int p  = p0 + ml;
        // invalid rows clamp to (0,0): phi = |H0 - H0| = 0, no guard needed
        const int ii = (p < NPAIR) ? (p / NNOD) : 0;
        const int jj = (p < NPAIR) ? (p % NNOD) : 0;
        const _Float16* rowI = &Hs[ii][0];
        const _Float16* rowJ = &Hs[jj][0];

        // A fragments: phi rows (M = lane%16); each lane needs two contiguous
        // 8-half K runs per k-step -> 4x ds_load_b128 + packed |a-b|.
        v16h aph[KS0];
#pragma unroll
        for (int ks = 0; ks < KS0; ++ks) {
            const int c1 = ks * 32 + 8 * half;
            v8h i1 = *(const v8h*)(rowI + c1);
            v8h j1 = *(const v8h*)(rowJ + c1);
            v8h i2 = *(const v8h*)(rowI + c1 + 16);
            v8h j2 = *(const v8h*)(rowJ + c1 + 16);
            v8h d1 = habs8(i1 - j1);
            v8h d2 = habs8(i2 - j2);
            aph[ks] = __builtin_shufflevector(d1, d2, 0, 1, 2, 3, 4, 5, 6, 7,
                                              8, 9, 10, 11, 12, 13, 14, 15);
        }

        // ---- stage 1: phi @ W0  (8 WMMA n-tiles, KS0 k-steps each) ----
#pragma unroll
        for (int nt = 0; nt < 8; ++nt) {
            v8f acc = {};
#pragma unroll
            for (int ks = 0; ks < KS0; ++ks) {
                v16h bf = *(const v16h*)(W0sw + ((size_t)(nt * KS0 + ks) * 32 + lane) * 16);
                acc = __builtin_amdgcn_wmma_f32_16x16x32_f16(
                    false, aph[ks], false, bf, (short)0, acc, false, false);
            }
            const int n = nt * 16 + ml;
            const float bn = b0s[n];
#pragma unroll
            for (int r = 0; r < 8; ++r) {
                float v = acc[r] + bn;                 // bias (BN folded)
                v = (v > 0.f) ? v : 0.01f * v;         // leaky_relu
                h0s[wave][r + 8 * half][n] = (_Float16)v;
            }
        }

        // ---- stage 2: h0 @ W1  (K=128 -> 4 k-steps, 4 n-tiles) ----
        v16h a1[4];
#pragma unroll
        for (int ks = 0; ks < 4; ++ks) {
            const int c1 = ks * 32 + 8 * half;
            v8h x1 = *(const v8h*)(&h0s[wave][ml][c1]);
            v8h x2 = *(const v8h*)(&h0s[wave][ml][c1 + 16]);
            a1[ks] = __builtin_shufflevector(x1, x2, 0, 1, 2, 3, 4, 5, 6, 7,
                                             8, 9, 10, 11, 12, 13, 14, 15);
        }
        v8f acc2[4];
#pragma unroll
        for (int nt = 0; nt < 4; ++nt) {
            v8f acc = {};
#pragma unroll
            for (int ks = 0; ks < 4; ++ks) {
                v16h bf = *(const v16h*)(W1sw + ((size_t)(nt * 4 + ks) * 32 + lane) * 16);
                acc = __builtin_amdgcn_wmma_f32_16x16x32_f16(
                    false, a1[ks], false, bf, (short)0, acc, false, false);
            }
            acc2[nt] = acc;
        }
#pragma unroll
        for (int nt = 0; nt < 4; ++nt) {
            const int n = nt * 16 + ml;
            const float bn = b1s[n];
#pragma unroll
            for (int r = 0; r < 8; ++r) {
                float v = acc2[nt][r] + bn;
                v = (v > 0.f) ? v : 0.01f * v;
                h0s[wave][r + 8 * half][n] = (_Float16)v;   // reuse buffer for h1
            }
        }
        // ---- stage 3: s = h1 . w2 + b2 (vectorized b128 reads) ----
        if (lane < 16) {
            const int pm = p0 + lane;
            if (pm < NPAIR) {
                float acc = w2s[64];
#pragma unroll
                for (int c = 0; c < 64; c += 8) {
                    v8h hv = *(const v8h*)(&h0s[wave][lane][c]);
#pragma unroll
                    for (int e = 0; e < 8; ++e) acc += (float)hv[e] * w2s[c + e];
                }
                sbuf[(size_t)b * NPAIR + pm] = acc;
            }
        }
    }
}

// ---------------------------------------------------------------------------
// Row softmax over j, degree, normalized Laplacian. One wave per episode.
// Optionally emit A (layer 2 -> d_out).
// ---------------------------------------------------------------------------
__global__ void __launch_bounds__(32) softmax_lap(const float* __restrict__ sbuf,
                                                  float* __restrict__ lap,
                                                  float* __restrict__ Aout) {
    int b = blockIdx.x, lane = threadIdx.x;
    __shared__ float A[NNOD][28];
    __shared__ float dinv[NNOD];
    if (lane < NNOD) {
        const float* srow = sbuf + (size_t)b * NPAIR + lane * NNOD;
        float mx = -1e30f;
        for (int j = 0; j < NNOD; ++j) mx = fmaxf(mx, srow[j]);
        float sum = 0.f, tmp[NNOD];
        for (int j = 0; j < NNOD; ++j) { float e = __expf(srow[j] - mx); tmp[j] = e; sum += e; }
        float inv = 1.f / sum, d = 0.f;
        for (int j = 0; j < NNOD; ++j) { float a = tmp[j] * inv; A[lane][j] = a; d += a; }
        dinv[lane] = rsqrtf(d + 1e-8f);
    }
    __syncthreads();
    for (int idx = lane; idx < NPAIR; idx += 32) {
        int i = idx / NNOD, j = idx % NNOD;
        float l = ((i == j) ? 1.f : 0.f) - dinv[i] * A[i][j] * dinv[j];
        lap[(size_t)b * NPAIR + idx] = l;
        if (Aout) Aout[(size_t)b * NPAIR + idx] = A[i][j];
    }
}

// ---------------------------------------------------------------------------
// Chebyshev K=3 graph conv. xt interleaves [x0,x1,x2] per feature (stack
// axis=-1 then reshape), W is (3*FIN, FOUT). One block per episode.
// ---------------------------------------------------------------------------
template <int FIN, int FOUT, bool LEAKY, bool HASB>
__global__ void __launch_bounds__(256) cheb(const float* __restrict__ Hbuf,
                                            const float* __restrict__ lap,
                                            const float* __restrict__ W,
                                            const float* __restrict__ bias,
                                            float* __restrict__ dst) {
    int b = blockIdx.x, t = threadIdx.x;
    __shared__ float L[NNOD][28];
    __shared__ float X0[NNOD][FIN + 1], X1[NNOD][FIN + 1], X2[NNOD][FIN + 1];
    for (int idx = t; idx < NPAIR; idx += 256) L[idx / NNOD][idx % NNOD] = lap[(size_t)b * NPAIR + idx];
    for (int idx = t; idx < NNOD * FIN; idx += 256) {
        int n = idx / FIN, f = idx % FIN;
        X0[n][f] = Hbuf[((size_t)b * NNOD + n) * HPITCH + f];
    }
    __syncthreads();
    for (int idx = t; idx < NNOD * FIN; idx += 256) {
        int n = idx / FIN, f = idx % FIN;
        float acc = 0.f;
        for (int j = 0; j < NNOD; ++j) acc += L[n][j] * X0[j][f];
        X1[n][f] = acc;
    }
    __syncthreads();
    for (int idx = t; idx < NNOD * FIN; idx += 256) {
        int n = idx / FIN, f = idx % FIN;
        float acc = 0.f;
        for (int j = 0; j < NNOD; ++j) acc += L[n][j] * X1[j][f];
        X2[n][f] = 2.f * acc - X0[n][f];
    }
    __syncthreads();
    for (int idx = t; idx < NNOD * FOUT; idx += 256) {
        int n = idx / FOUT, c = idx % FOUT;
        float acc = HASB ? bias[c] : 0.f;
        for (int f = 0; f < FIN; ++f) {
            acc += X0[n][f] * W[(f * 3 + 0) * FOUT + c];
            acc += X1[n][f] * W[(f * 3 + 1) * FOUT + c];
            acc += X2[n][f] * W[(f * 3 + 2) * FOUT + c];
        }
        if (LEAKY) acc = (acc > 0.f) ? acc : 0.01f * acc;
        if (FOUT == 5) {
            if (n == 0) dst[(size_t)b * 5 + c] = acc;   // only query node needed
        } else {
            dst[((size_t)b * NNOD + n) * HPITCH + FIN + c] = acc;  // concat onto h
        }
    }
}

__global__ void lsm(const float* __restrict__ out0, float* __restrict__ dout) {
    int b = blockIdx.x * blockDim.x + threadIdx.x;
    if (b < BSZ) {
        float v[5], mx = -1e30f;
#pragma unroll
        for (int c = 0; c < 5; ++c) { v[c] = out0[b * 5 + c]; mx = fmaxf(mx, v[c]); }
        float s = 0.f;
#pragma unroll
        for (int c = 0; c < 5; ++c) s += __expf(v[c] - mx);
        float ls = __logf(s) + mx;
#pragma unroll
        for (int c = 0; c < 5; ++c) dout[b * 5 + c] = v[c] - ls;
    }
}

// ---------------------------------------------------------------------------
extern "C" void kernel_launch(void* const* d_in, const int* in_sizes, int n_in,
                              void* d_out, int out_size, void* d_ws, size_t ws_size,
                              hipStream_t stream) {
    (void)in_sizes; (void)n_in; (void)out_size; (void)ws_size;
    int i = 0;
    const float* x  = (const float*)d_in[i++];
    const float* xi = (const float*)d_in[i++];
    const float* oh = (const float*)d_in[i++];
    // cnn params
    const float* c1w = (const float*)d_in[i++]; const float* c1b = (const float*)d_in[i++];
    const float* g1  = (const float*)d_in[i++]; const float* be1 = (const float*)d_in[i++];
    const float* m1  = (const float*)d_in[i++]; const float* v1  = (const float*)d_in[i++];
    const float* c2w = (const float*)d_in[i++]; const float* c2b = (const float*)d_in[i++];
    const float* g2  = (const float*)d_in[i++]; const float* be2 = (const float*)d_in[i++];
    const float* m2  = (const float*)d_in[i++]; const float* v2  = (const float*)d_in[i++];
    const float* c3w = (const float*)d_in[i++]; const float* c3b = (const float*)d_in[i++];
    const float* g3  = (const float*)d_in[i++]; const float* be3 = (const float*)d_in[i++];
    const float* m3  = (const float*)d_in[i++]; const float* v3  = (const float*)d_in[i++];
    const float* f1w = (const float*)d_in[i++]; const float* f1b = (const float*)d_in[i++];
    const float* g4  = (const float*)d_in[i++]; const float* be4 = (const float*)d_in[i++];
    const float* m4  = (const float*)d_in[i++]; const float* v4  = (const float*)d_in[i++];
    const float* f2w = (const float*)d_in[i++]; const float* f2b = (const float*)d_in[i++];
    // adj params: 3 layers x 14 entries
    const float* aw0[3]; const float* ab0[3]; const float* ag0[3]; const float* abe0[3];
    const float* am0[3]; const float* av0[3];
    const float* aw1[3]; const float* ab1[3]; const float* ag1[3]; const float* abe1[3];
    const float* am1[3]; const float* av1[3];
    const float* aw2[3]; const float* ab2[3];
    for (int l = 0; l < 3; ++l) {
        aw0[l]=(const float*)d_in[i++]; ab0[l]=(const float*)d_in[i++];
        ag0[l]=(const float*)d_in[i++]; abe0[l]=(const float*)d_in[i++];
        am0[l]=(const float*)d_in[i++]; av0[l]=(const float*)d_in[i++];
        aw1[l]=(const float*)d_in[i++]; ab1[l]=(const float*)d_in[i++];
        ag1[l]=(const float*)d_in[i++]; abe1[l]=(const float*)d_in[i++];
        am1[l]=(const float*)d_in[i++]; av1[l]=(const float*)d_in[i++];
        aw2[l]=(const float*)d_in[i++]; ab2[l]=(const float*)d_in[i++];
    }
    const float* chW0 = (const float*)d_in[i++]; const float* chb0 = (const float*)d_in[i++];
    const float* chW1 = (const float*)d_in[i++]; const float* chb1 = (const float*)d_in[i++];
    const float* chW2 = (const float*)d_in[i++];

    float* out = (float*)d_out;      // [B*5 logsoftmax][B*676 A]
    float* ws = (float*)d_ws;
    float* H    = ws;                                  // B*26*96
    float* S    = ws + (size_t)BSZ * NNOD * HPITCH;    // B*676 scores
    float* LAP  = S + (size_t)BSZ * NPAIR;             // B*676
    float* OUT0 = LAP + (size_t)BSZ * NPAIR;           // B*5
    float* FOLD = OUT0 + (size_t)BSZ * 5;              // 256
    float* ADJ  = FOLD + 256;
    const size_t LSTR = 10560;                         // per adj-layer region (floats)

    prep_cnn<<<1, 32, 0, stream>>>(c1b, g1, be1, m1, v1, c2b, g2, be2, m2, v2,
                                   c3b, g3, be3, m3, v3, f1b, g4, be4, m4, v4, FOLD);
    cnn_fwd<<<BSZ * NNOD, 64, 0, stream>>>(x, xi, oh, c1w, c2w, c3w, f1w, f2w, FOLD, f2b, H);

    // ----- layer 0: F=23 (1 k-step) -----
    {
        float* R = ADJ; float* bias0 = R; float* bias1 = R + 128; float* w2f = R + 192;
        _Float16* W0sw = (_Float16*)(R + 272); _Float16* W1sw = (_Float16*)(R + 272 + 6144);
        prep_adj<23, 1><<<1, 256, 0, stream>>>(aw0[0], ab0[0], ag0[0], abe0[0], am0[0], av0[0],
                                               aw1[0], ab1[0], ag1[0], abe1[0], am1[0], av1[0],
                                               aw2[0], ab2[0], W0sw, bias0, W1sw, bias1, w2f);
        adj_mlp<23, 1><<<BSZ, 256, 0, stream>>>(H, W0sw, bias0, W1sw, bias1, w2f, S);
        softmax_lap<<<BSZ, 32, 0, stream>>>(S, LAP, nullptr);
        cheb<23, 32, true, true><<<BSZ, 256, 0, stream>>>(H, LAP, chW0, chb0, H);
    }
    // ----- layer 1: F=55 (2 k-steps) -----
    {
        float* R = ADJ + LSTR; float* bias0 = R; float* bias1 = R + 128; float* w2f = R + 192;
        _Float16* W0sw = (_Float16*)(R + 272); _Float16* W1sw = (_Float16*)(R + 272 + 6144);
        prep_adj<55, 2><<<1, 256, 0, stream>>>(aw0[1], ab0[1], ag0[1], abe0[1], am0[1], av0[1],
                                               aw1[1], ab1[1], ag1[1], abe1[1], am1[1], av1[1],
                                               aw2[1], ab2[1], W0sw, bias0, W1sw, bias1, w2f);
        adj_mlp<55, 2><<<BSZ, 256, 0, stream>>>(H, W0sw, bias0, W1sw, bias1, w2f, S);
        softmax_lap<<<BSZ, 32, 0, stream>>>(S, LAP, nullptr);
        cheb<55, 32, true, true><<<BSZ, 256, 0, stream>>>(H, LAP, chW1, chb1, H);
    }
    // ----- layer 2: F=87 (3 k-steps), emit A and logits -----
    {
        float* R = ADJ + 2 * LSTR; float* bias0 = R; float* bias1 = R + 128; float* w2f = R + 192;
        _Float16* W0sw = (_Float16*)(R + 272); _Float16* W1sw = (_Float16*)(R + 272 + 6144);
        prep_adj<87, 3><<<1, 256, 0, stream>>>(aw0[2], ab0[2], ag0[2], abe0[2], am0[2], av0[2],
                                               aw1[2], ab1[2], ag1[2], abe1[2], am1[2], av1[2],
                                               aw2[2], ab2[2], W0sw, bias0, W1sw, bias1, w2f);
        adj_mlp<87, 3><<<BSZ, 256, 0, stream>>>(H, W0sw, bias0, W1sw, bias1, w2f, S);
        softmax_lap<<<BSZ, 32, 0, stream>>>(S, LAP, out + (size_t)BSZ * 5);
        cheb<87, 5, false, false><<<BSZ, 256, 0, stream>>>(H, LAP, chW2, nullptr, OUT0);
    }
    lsm<<<(BSZ + 63) / 64, 64, 0, stream>>>(OUT0, out);
}
